// NonLocalFeatureFusionPart_23304492548791
// MI455X (gfx1250) — compile-verified, matrix operand order checked by reading, not run on hardware
//
#include <hip/hip_runtime.h>

typedef _Float16 half_t;
typedef __attribute__((ext_vector_type(16))) _Float16 v16h;
typedef __attribute__((ext_vector_type(8)))  float    v8f;

#define N_TOK 4096
#define KC    64
#define VC    64
#define CH    256
#define CL    512
#define NB    4

// ---------------------------------------------------------------------------
// CDNA5 async global->LDS copy (ASYNCcnt-tracked), with sync fallback.
// ---------------------------------------------------------------------------
#if defined(__has_builtin)
#  if __has_builtin(__builtin_amdgcn_global_load_async_to_lds_b128) && \
      __has_builtin(__builtin_amdgcn_s_wait_asynccnt)
#    define USE_ASYNC_COPY 1
#  endif
#endif
#ifndef USE_ASYNC_COPY
#  define USE_ASYNC_COPY 0
#endif

#if USE_ASYNC_COPY
typedef int v4i_vs __attribute__((vector_size(16)));
typedef __attribute__((address_space(1))) v4i_vs* gp128;
typedef __attribute__((address_space(3))) v4i_vs* lp128;
__device__ __forceinline__ void lds_copy16(const void* g, void* l) {
  // generic LDS ptr -> as(3): ISA 10.2, keep the low 32 address bits
  __builtin_amdgcn_global_load_async_to_lds_b128(
      (gp128)(unsigned long long)g,
      (lp128)(unsigned int)(unsigned long long)l, 0, 0);
}
__device__ __forceinline__ void lds_copy_wait() {
  __builtin_amdgcn_s_wait_asynccnt(0);
}
#else
__device__ __forceinline__ void lds_copy16(const void* g, void* l) {
  *(uint4*)l = *(const uint4*)g;
}
__device__ __forceinline__ void lds_copy_wait() {}
#endif

#if defined(__has_builtin) && __has_builtin(__builtin_amdgcn_rcpf)
#  define RCPF(x) __builtin_amdgcn_rcpf(x)
#else
#  define RCPF(x) (1.0f / (x))
#endif

__device__ __forceinline__ v8f wmma32(v16h a, v16h b, v8f c) {
  // D = A(16x32 f16) x B(32x16 f16) + C(16x16 f32)
  return __builtin_amdgcn_wmma_f32_16x16x32_f16(
      /*neg_a=*/false, a, /*neg_b=*/false, b,
      /*c_mod=*/(short)0, c, /*reuse_a=*/false, /*reuse_b=*/false);
}

// 16x32 f16 A-fragment (also serves B when B is stored transposed, [n][k])
// from a row-major [row][k] buffer with leading dimension ld (halves).
// Lanes 0-15: row M=lane, K slots {0..7,16..23}; lanes 16-31: row M=lane-16,
// K slots {8..15,24..31}.
__device__ __forceinline__ v16h load_frag(const half_t* p0, int row0, int k0, int ld) {
  const int lane = threadIdx.x & 31;
  const half_t* p = p0 + (size_t)(row0 + (lane & 15)) * ld + k0 + ((lane >> 4) << 3);
  v16h a;
#pragma unroll
  for (int i = 0; i < 8; ++i) { a[i] = p[i]; a[8 + i] = p[16 + i]; }
  return a;
}

// ---------------------------------------------------------------------------
// Kernel 1: projection  y[j][n] = sum_c X[b][c][n]*W[j][c]+bias[j] (f16 out)
// TRANS=0: Y[b][n][j] (token-major, used for q,k)
// TRANS=1: Y[b][j][n] (channel-major, used for v -> direct V^T A-operand)
// Tile: 128 tokens x 64 outputs per workgroup (256 threads = 8 waves).
// ---------------------------------------------------------------------------
template <int C, bool TRANS>
__global__ __launch_bounds__(256)
void qkv_proj_kernel(const float* __restrict__ X, const float* __restrict__ W,
                     const float* __restrict__ bias, half_t* __restrict__ Y) {
  __shared__ half_t Xs[128 * 40];   // [token][c-chunk], ld=40 (pad)
  __shared__ half_t Ws[64 * 40];    // [j][c-chunk]  == B^T layout
  const int b  = blockIdx.y;
  const int n0 = blockIdx.x * 128;
  const int t  = threadIdx.x;
  const int w  = t >> 5;
  const float* Xb = X + (size_t)b * C * N_TOK;

  v8f acc[4] = {};
  for (int c0 = 0; c0 < C; c0 += 32) {
    __syncthreads();
#pragma unroll
    for (int i = 0; i < 16; ++i) {             // 32 channels x 128 tokens
      int e = t + 256 * i;
      int k = e >> 7, m = e & 127;
      Xs[m * 40 + k] = (half_t)Xb[(size_t)(c0 + k) * N_TOK + n0 + m];
    }
#pragma unroll
    for (int i = 0; i < 8; ++i) {              // 64 j x 32 c
      int e = t + 256 * i;
      int j = e >> 5, k = e & 31;
      Ws[j * 40 + k] = (half_t)W[(size_t)j * C + c0 + k];
    }
    __syncthreads();
    v16h a = load_frag(Xs, 16 * w, 0, 40);
#pragma unroll
    for (int jt = 0; jt < 4; ++jt) {
      v16h bf = load_frag(Ws, 16 * jt, 0, 40);
      acc[jt] = wmma32(a, bf, acc[jt]);
    }
  }
  const int lane = t & 31;
  const int hi   = lane >> 4;
#pragma unroll
  for (int jt = 0; jt < 4; ++jt) {
    int j = 16 * jt + (lane & 15);
    float bj = bias[j];
    if (TRANS) {
      // C-layout rows are consecutive tokens -> one b128 store per tile
      half_t tmp[8];
#pragma unroll
      for (int r = 0; r < 8; ++r) tmp[r] = (half_t)(acc[jt][r] + bj);
      int nbase = n0 + 16 * w + 8 * hi;
      *(uint4*)&Y[((size_t)b * 64 + j) * N_TOK + nbase] = *(const uint4*)tmp;
    } else {
#pragma unroll
      for (int r = 0; r < 8; ++r) {
        int m = 16 * w + r + 8 * hi;
        Y[((size_t)b * N_TOK + n0 + m) * 64 + j] = (half_t)(acc[jt][r] + bj);
      }
    }
  }
}

// ---------------------------------------------------------------------------
// Kernel 2: flash attention, transposed-score form. 8 waves/WG, 16 queries/wave.
//  S^T = K_tile x Q^T : C-result layout == B-fragment layout of P^T, so the
//  exp() happens entirely in registers and feeds ctx^T = V^T x P^T directly.
//  Per-lane scalar softmax stats (query = lane&15), one xor-16 shuffle.
// ---------------------------------------------------------------------------
__global__ __launch_bounds__(256)
void attn_kernel(const half_t* __restrict__ Q, const half_t* __restrict__ K,
                 const half_t* __restrict__ Vt, half_t* __restrict__ CTX) {
  __shared__ half_t Ks[64 * 72];        // [key][kc]  : A-operand of S^T
  __shared__ half_t Vs[64 * 72];        // [vc][key]  : A-operand of ctx^T
  const int b    = blockIdx.y;
  const int qb   = blockIdx.x * 128;
  const int t    = threadIdx.x;
  const int w    = t >> 5;
  const int lane = t & 31;
  const int hi   = lane >> 4;

  // Q as B-operand (B^T row-major = q[token][kc] global layout)
  const half_t* qp = Q + ((size_t)b * N_TOK + qb + 16 * w) * 64;
  v16h qf0 = load_frag(qp, 0, 0, 64);
  v16h qf1 = load_frag(qp, 0, 32, 64);

  v8f acc[4] = {};
  float mrow = -1e30f, lrow = 0.f;      // per-lane scalars (one query per lane)

  const half_t* Kb = K  + (size_t)b * N_TOK * 64;
  const half_t* Vb = Vt + (size_t)b * 64 * N_TOK;

  for (int kb = 0; kb < N_TOK; kb += 64) {
    __syncthreads();                    // all waves done with prev Ks/Vs
    // async-stage K block [key][kc] and V^T block [vc][key]: 512+512 chunks
#pragma unroll
    for (int i = 0; i < 2; ++i) {
      int ci = t + 256 * i;
      int row = ci >> 3, g = ci & 7;
      const half_t* gk = Kb + (size_t)(kb + row) * 64 + g * 8;
      const half_t* gv = Vb + (size_t)row * N_TOK + kb + g * 8;
      lds_copy16(gk, &Ks[row * 72 + g * 8]);
      lds_copy16(gv, &Vs[row * 72 + g * 8]);
      if (kb + 64 < N_TOK) {            // stream next block toward L2
        __builtin_prefetch(gk + 64 * 64, 0, 1);
        __builtin_prefetch(gv + 64, 0, 1);
      }
    }
    lds_copy_wait();                    // own ASYNCcnt -> 0
    __syncthreads();                    // everyone's staging visible

    // S^T tiles: key-tile kt (M) x 16 queries (N), KC=64 = two k=32 chunks
    v8f s[4];
#pragma unroll
    for (int kt = 0; kt < 4; ++kt) {
      v8f z = {};
      z = wmma32(load_frag(Ks, 16 * kt, 0, 72), qf0, z);
      z = wmma32(load_frag(Ks, 16 * kt, 32, 72), qf1, z);
      s[kt] = z;
    }

    // block max over the 32 keys this lane holds, then fold in other half
    float bm = s[0][0];
#pragma unroll
    for (int kt = 0; kt < 4; ++kt)
#pragma unroll
      for (int r = 0; r < 8; ++r) bm = fmaxf(bm, s[kt][r]);
    bm = fmaxf(bm, __shfl_xor(bm, 16, 32));
    float nm = fmaxf(mrow, bm);
    float sc = __expf(mrow - nm);
    mrow = nm;

    // exp in registers; pack straight into P^T B-fragments:
    //  chunk0 keys 0..31 = tiles {0,1}, chunk1 keys 32..63 = tiles {2,3}
    v16h pf0, pf1;
    float bs = 0.f;
#pragma unroll
    for (int r = 0; r < 8; ++r) {
      float p0 = __expf(s[0][r] - nm);
      float p1 = __expf(s[1][r] - nm);
      float p2 = __expf(s[2][r] - nm);
      float p3 = __expf(s[3][r] - nm);
      bs += (p0 + p1) + (p2 + p3);
      pf0[r] = (half_t)p0; pf0[8 + r] = (half_t)p1;
      pf1[r] = (half_t)p2; pf1[8 + r] = (half_t)p3;
    }
    bs += __shfl_xor(bs, 16, 32);
    lrow = lrow * sc + bs;
#pragma unroll
    for (int vt = 0; vt < 4; ++vt)
#pragma unroll
      for (int r = 0; r < 8; ++r) acc[vt][r] *= sc;

    // ctx^T (vc x query) += V^T x P^T
#pragma unroll
    for (int vt = 0; vt < 4; ++vt) {
      acc[vt] = wmma32(load_frag(Vs, 16 * vt, 0, 72), pf0, acc[vt]);
      acc[vt] = wmma32(load_frag(Vs, 16 * vt, 32, 72), pf1, acc[vt]);
    }
  }

  // normalize + store: per lane vc = 16*vt + 8*hi + r are consecutive -> b128
  float inv = RCPF(lrow);
  const int query = lane & 15;
#pragma unroll
  for (int vt = 0; vt < 4; ++vt) {
    half_t tmp[8];
#pragma unroll
    for (int r = 0; r < 8; ++r) tmp[r] = (half_t)(acc[vt][r] * inv);
    *(uint4*)&CTX[((size_t)b * N_TOK + qb + 16 * w + query) * 64 + 16 * vt + 8 * hi] =
        *(const uint4*)tmp;
  }
}

// ---------------------------------------------------------------------------
// Kernel 3: out[b][j][n] = sum_v Wo[j][v]*ctx[b][n][v] + bo[j] + x_high[b][j][n]
// Tile: 128 tokens x 64 output channels (grid.z selects the channel tile).
// ---------------------------------------------------------------------------
__global__ __launch_bounds__(256)
void out_proj_kernel(const half_t* __restrict__ CTX, const float* __restrict__ Wo,
                     const float* __restrict__ bo, const float* __restrict__ XH,
                     float* __restrict__ OUT) {
  __shared__ half_t As[128 * 72];   // [token][vc]
  __shared__ half_t Bs[64 * 72];    // [j][vc] == B^T
  const int b  = blockIdx.y;
  const int n0 = blockIdx.x * 128;
  const int j0 = blockIdx.z * 64;
  const int t  = threadIdx.x;
  const int w  = t >> 5;
#pragma unroll
  for (int i = 0; i < 4; ++i) {       // 1024 16B chunks of ctx tile (async)
    int ci = t + 256 * i;
    int m = ci >> 3, g = ci & 7;
    lds_copy16(CTX + ((size_t)b * N_TOK + n0 + m) * 64 + g * 8,
               &As[m * 72 + g * 8]);
  }
#pragma unroll
  for (int i = 0; i < 16; ++i) {      // 64 j x 64 v, f32 -> f16
    int e = t + 256 * i;
    int j = e >> 6, v = e & 63;
    Bs[j * 72 + v] = (half_t)Wo[(size_t)(j0 + j) * VC + v];
  }
  lds_copy_wait();
  __syncthreads();
  v16h a0 = load_frag(As, 16 * w, 0, 72);
  v16h a1 = load_frag(As, 16 * w, 32, 72);
  v8f acc[4] = {};
#pragma unroll
  for (int jt = 0; jt < 4; ++jt) {
    acc[jt] = wmma32(a0, load_frag(Bs, 16 * jt, 0, 72), acc[jt]);
    acc[jt] = wmma32(a1, load_frag(Bs, 16 * jt, 32, 72), acc[jt]);
  }
  const int lane = t & 31;
  const int hi   = lane >> 4;
#pragma unroll
  for (int jt = 0; jt < 4; ++jt) {
    int j = j0 + 16 * jt + (lane & 15);
    float bj = bo[j];
#pragma unroll
    for (int r = 0; r < 8; ++r) {
      int m = 16 * w + r + 8 * hi;
      size_t idx = ((size_t)b * CH + j) * N_TOK + n0 + m;
      OUT[idx] = acc[jt][r] + bj + XH[idx];
    }
  }
}

// ---------------------------------------------------------------------------
extern "C" void kernel_launch(void* const* d_in, const int* in_sizes, int n_in,
                              void* d_out, int out_size, void* d_ws, size_t ws_size,
                              hipStream_t stream) {
  const float* x_low  = (const float*)d_in[0];
  const float* x_high = (const float*)d_in[1];
  const float* Wq = (const float*)d_in[2];
  const float* bq = (const float*)d_in[3];
  const float* Wk = (const float*)d_in[4];
  const float* bk = (const float*)d_in[5];
  const float* Wv = (const float*)d_in[6];
  const float* bv = (const float*)d_in[7];
  const float* Wo = (const float*)d_in[8];
  const float* bo = (const float*)d_in[9];
  float* out = (float*)d_out;

  // workspace: q, k, v^T, ctx each B*N*64 f16 = 2 MB  (8 MB total)
  half_t* qf = (half_t*)d_ws;
  half_t* kf = qf + (size_t)NB * N_TOK * 64;
  half_t* vf = kf + (size_t)NB * N_TOK * 64;   // stored [b][vc][n]
  half_t* cf = vf + (size_t)NB * N_TOK * 64;

  dim3 pgrid(N_TOK / 128, NB);
  qkv_proj_kernel<CH, false><<<pgrid, dim3(256), 0, stream>>>(x_high, Wq, bq, qf);
  qkv_proj_kernel<CL, false><<<pgrid, dim3(256), 0, stream>>>(x_low,  Wk, bk, kf);
  qkv_proj_kernel<CL, true ><<<pgrid, dim3(256), 0, stream>>>(x_low,  Wv, bv, vf);

  attn_kernel<<<dim3(N_TOK / 128, NB), dim3(256), 0, stream>>>(qf, kf, vf, cf);

  out_proj_kernel<<<dim3(N_TOK / 128, NB, CH / 64), dim3(256), 0, stream>>>(
      cf, Wo, bo, x_high, out);
}
